// DoubleAttention_71021579206932
// MI455X (gfx1250) — compile-verified, matrix operand order checked by reading, not run on hardware
//
#include <hip/hip_runtime.h>
#include <hip/hip_bf16.h>

typedef __attribute__((ext_vector_type(16))) _Float16 v16h;
typedef __attribute__((ext_vector_type(8)))  _Float16 v8h;
typedef __attribute__((ext_vector_type(8)))  float    v8f;
typedef __attribute__((ext_vector_type(4)))  int      v4i;

#define DIM 2048
#define HEADS 16
#define HD 128
#define SEQ 2048   // concatenated sequence length per batch
#define CLEN 256
#define XLEN 1792

// ---------------------------------------------------------------------------
// CDNA5 async global->LDS copy (guarded; falls back to synchronous copy)
// Builtin prototype (probe-confirmed by round-2 diagnostic):
//   void __builtin_amdgcn_global_load_async_to_lds_b128(
//       global int4* src, local int4* dst, imm int offset, imm int cpol)
// ---------------------------------------------------------------------------
#if defined(__has_builtin)
#if __has_builtin(__builtin_amdgcn_global_load_async_to_lds_b128)
#define HAVE_ASYNC 1
#endif
#endif
#ifndef HAVE_ASYNC
#define HAVE_ASYNC 0
#endif

#if HAVE_ASYNC
typedef __attribute__((address_space(1))) v4i gv4i_t;
typedef __attribute__((address_space(3))) v4i sv4i_t;
#endif

__device__ __forceinline__ void copy_b128_to_lds(const _Float16* g, _Float16* l) {
#if HAVE_ASYNC
  __builtin_amdgcn_global_load_async_to_lds_b128((gv4i_t*)g, (sv4i_t*)l, 0, 0);
#else
  *(v8h*)l = *(const v8h*)g;
#endif
}

__device__ __forceinline__ void wait_async0() {
#if HAVE_ASYNC
#if __has_builtin(__builtin_amdgcn_s_wait_asynccnt)
  __builtin_amdgcn_s_wait_asynccnt(0);
#else
  asm volatile("s_wait_asynccnt 0" ::: "memory");
#endif
#endif
}

// ---------------------------------------------------------------------------
// WMMA helpers
// ---------------------------------------------------------------------------
__device__ __forceinline__ v8f wmma16(v16h a, v16h b, v8f c) {
  // D = A(16x32 f16) * B(32x16 f16) + C(16x16 f32)
  return __builtin_amdgcn_wmma_f32_16x16x32_f16(
      /*neg_a=*/false, a, /*neg_b=*/false, b,
      /*c_mod=*/(short)0, c, /*reuse_a=*/false, /*reuse_b=*/false);
}

// Load a 16x32 f16 operand fragment from a row-major matrix (row stride = ld
// elements).  Per ISA 7.12.2: lane l holds row (l&15); halves i=0..7 cover
// k = 8*(l>>4) + i, halves i=8..15 cover k = 8*(l>>4) + 16 + i-8.
// Works for global or LDS sources (addrspace inferred after inlining).
__device__ __forceinline__ v16h load_frag(const _Float16* base, int ld) {
  const int l  = threadIdx.x & 31;
  const int r  = l & 15;
  const int hi = (l >> 4) * 8;
  const _Float16* p = base + (size_t)r * ld + hi;
  v8h a = *(const v8h*)(p);
  v8h b = *(const v8h*)(p + 16);
  v16h o;
#pragma unroll
  for (int i = 0; i < 8; ++i) { o[i] = a[i]; o[i + 8] = b[i]; }
  return o;
}

// ---------------------------------------------------------------------------
// Elementwise fp32 -> f16
// ---------------------------------------------------------------------------
__global__ void f32_to_f16_kernel(const float* __restrict__ in,
                                  _Float16* __restrict__ out, int n) {
  int i = blockIdx.x * 256 + threadIdx.x;
  if (i < n) out[i] = (_Float16)in[i];
}

// Transpose + convert one 2048x2048 fp32 weight matrix W[k][n] into
// f16 Wt[n][k] (so B fragments read contiguous K runs).
__global__ void weight_T_kernel(const float* __restrict__ W,
                                _Float16* __restrict__ Wt) {
  __shared__ float tile[32][33];
  const int n0 = blockIdx.x * 32;
  const int k0 = blockIdx.y * 32;
  const int tx = threadIdx.x;
#pragma unroll
  for (int r = threadIdx.y; r < 32; r += 8)
    tile[r][tx] = W[(size_t)(k0 + r) * DIM + n0 + tx];
  __syncthreads();
#pragma unroll
  for (int r = threadIdx.y; r < 32; r += 8)
    Wt[(size_t)(n0 + r) * DIM + k0 + tx] = (_Float16)tile[tx][r];
}

// ---------------------------------------------------------------------------
// GEMM: C[M x 2048] (fp32) = A[M x 2048] (f16, lda=2048) * Wt^T
// Wt is the transposed weight, f16 [n][k].  One wave computes a 16x64 tile.
// Software-pipelined: fragments for k+32 are loaded while WMMAs consume k.
// grid = (M/16, 32), block = 32 (one wave).
// ---------------------------------------------------------------------------
__global__ __launch_bounds__(32)
void gemm_f16_kernel(const _Float16* __restrict__ A,
                     const _Float16* __restrict__ Bt,
                     float* __restrict__ C, int M) {
  const int m0 = blockIdx.x * 16;
  const int n0 = blockIdx.y * 64;
  const _Float16* Arow = A + (size_t)m0 * DIM;

  v8f acc[4];
#pragma unroll
  for (int t = 0; t < 4; ++t)
#pragma unroll
    for (int j = 0; j < 8; ++j) acc[t][j] = 0.f;

  v16h a_cur = load_frag(Arow, DIM);
  v16h b_cur[4];
#pragma unroll
  for (int t = 0; t < 4; ++t)
    b_cur[t] = load_frag(Bt + (size_t)(n0 + t * 16) * DIM, DIM);

#pragma unroll 2
  for (int k0 = 0; k0 < DIM; k0 += 32) {
    const int k1 = (k0 + 32) & (DIM - 1);   // wraps to 0 on last iter (dead)
    v16h a_nxt = load_frag(Arow + k1, DIM);
    v16h b_nxt[4];
#pragma unroll
    for (int t = 0; t < 4; ++t)
      b_nxt[t] = load_frag(Bt + (size_t)(n0 + t * 16) * DIM + k1, DIM);
#pragma unroll
    for (int t = 0; t < 4; ++t) acc[t] = wmma16(a_cur, b_cur[t], acc[t]);
    a_cur = a_nxt;
#pragma unroll
    for (int t = 0; t < 4; ++t) b_cur[t] = b_nxt[t];
  }

  const int l = threadIdx.x & 31;
  const int n = l & 15;
  const int rb = (l >> 4) * 8;
#pragma unroll
  for (int t = 0; t < 4; ++t)
#pragma unroll
    for (int j = 0; j < 8; ++j)
      C[(size_t)(m0 + rb + j) * DIM + n0 + t * 16 + n] = acc[t][j];
}

// ---------------------------------------------------------------------------
// Fused LayerNorm (fp32, over 2048) + RoPE, output f16.
// src: [2, 2048, 2048]; row = b*2048+s.  s<256 -> stream c (params *_c,
// position s); s>=256 -> stream x (params *_x, position s-256).
// ---------------------------------------------------------------------------
__global__ void ln_rope_kernel(const float* __restrict__ src,
                               const float* __restrict__ w_c, const float* __restrict__ b_c,
                               const float* __restrict__ w_x, const float* __restrict__ b_x,
                               const float* __restrict__ fre, const float* __restrict__ fim,
                               _Float16* __restrict__ dst) {
  const int row = blockIdx.x;
  const int s = row & (SEQ - 1);
  const bool is_c = (s < CLEN);
  const int pos = is_c ? s : (s - CLEN);
  const float* w = is_c ? w_c : w_x;
  const float* bb = is_c ? b_c : b_x;
  const float* p = src + (size_t)row * DIM;

  float sum = 0.f, sum2 = 0.f;
  for (int i = threadIdx.x; i < DIM; i += 256) {
    float v = p[i];
    sum += v; sum2 += v * v;
  }
#pragma unroll
  for (int m = 16; m >= 1; m >>= 1) {
    sum  += __shfl_xor(sum, m, 32);
    sum2 += __shfl_xor(sum2, m, 32);
  }
  __shared__ float s1[8], s2[8];
  __shared__ float mu_s, rs_s;
  if ((threadIdx.x & 31) == 0) { s1[threadIdx.x >> 5] = sum; s2[threadIdx.x >> 5] = sum2; }
  __syncthreads();
  if (threadIdx.x == 0) {
    float a = 0.f, b2 = 0.f;
#pragma unroll
    for (int i = 0; i < 8; ++i) { a += s1[i]; b2 += s2[i]; }
    float mu = a * (1.f / DIM);
    float var = b2 * (1.f / DIM) - mu * mu;
    mu_s = mu;
    rs_s = rsqrtf(var + 1e-5f);
  }
  __syncthreads();
  const float mu = mu_s, rs = rs_s;

  for (int pr = threadIdx.x; pr < DIM / 2; pr += 256) {
    int c0 = pr * 2, c1 = c0 + 1;
    int dp = pr & (HD / 2 - 1);     // pair index within head
    float xr = (p[c0] - mu) * rs * w[c0] + bb[c0];
    float xi = (p[c1] - mu) * rs * w[c1] + bb[c1];
    float fr = fre[pos * (HD / 2) + dp];
    float fi = fim[pos * (HD / 2) + dp];
    dst[(size_t)row * DIM + c0] = (_Float16)(xr * fr - xi * fi);
    dst[(size_t)row * DIM + c1] = (_Float16)(xr * fi + xi * fr);
  }
}

// V: fp32 [b, s, h, d] -> f16 transposed [b, h, d, s]
__global__ void v_transpose_kernel(const float* __restrict__ v,
                                   _Float16* __restrict__ vt) {
  size_t i = (size_t)blockIdx.x * 256 + threadIdx.x;  // over 2*2048*2048
  int d = i & (HD - 1);
  int h = (i >> 7) & (HEADS - 1);
  int s = (i >> 11) & (SEQ - 1);
  int b = i >> 22;
  vt[(((size_t)(b * HEADS + h) * HD + d) * SEQ) + s] = (_Float16)v[i];
}

// ---------------------------------------------------------------------------
// Flash attention, 4 waves per workgroup (64 query rows), K/V tiles staged
// into LDS with double-buffered async global->LDS copies shared by all waves.
// Q/K: f16 [b, s, h*128]; Vt: f16 [b, h, d, s]; out: f16 [b, s, h*128].
// grid = (SEQ/64, HEADS, 2), block = 128.
// ---------------------------------------------------------------------------
__global__ __launch_bounds__(128)
void attn_kernel(const _Float16* __restrict__ Q, const _Float16* __restrict__ K,
                 const _Float16* __restrict__ Vt, _Float16* __restrict__ O) {
  const int qb = blockIdx.x, h = blockIdx.y, b = blockIdx.z;
  const size_t base = (size_t)b * SEQ * DIM;
  const int wave = threadIdx.x >> 5;
  const _Float16* Qp = Q + base + (size_t)(qb * 64 + wave * 16) * DIM + h * HD;
  const _Float16* Kp = K + base + h * HD;
  const _Float16* Vp = Vt + (size_t)(b * HEADS + h) * HD * SEQ;

  __shared__ __attribute__((aligned(16))) _Float16 kbuf[2][32][128];  // [key][d]
  __shared__ __attribute__((aligned(16))) _Float16 vbuf[2][128][32];  // [d][key]
  __shared__ __attribute__((aligned(16))) _Float16 pbuf[4][16 * 32];  // per wave

  const int l = threadIdx.x & 31;
  const int n = l & 15;
  const int rb = (l >> 4) * 8;

  v16h qf[4];
#pragma unroll
  for (int dk = 0; dk < 4; ++dk) qf[dk] = load_frag(Qp + dk * 32, DIM);

  v8f o[8];
  float mrow[8], lrow[8];
#pragma unroll
  for (int t = 0; t < 8; ++t)
#pragma unroll
    for (int j = 0; j < 8; ++j) o[t][j] = 0.f;
#pragma unroll
  for (int j = 0; j < 8; ++j) { mrow[j] = -3.0e38f; lrow[j] = 0.f; }

  // stage one 32-key tile of K (32x128) and V (128x32) into LDS buffer `buf`
  auto stage = [&](int buf, int j0) {
    const int t = threadIdx.x;  // 0..127
#pragma unroll
    for (int r = 0; r < 4; ++r) {
      int idx = t + r * 128;
      int row = idx >> 4, ch = idx & 15;        // K: 32 rows x 16 chunks of 8h
      copy_b128_to_lds(Kp + (size_t)(j0 + row) * DIM + ch * 8,
                       &kbuf[buf][row][ch * 8]);
      int d = idx >> 2, cv = idx & 3;           // V: 128 rows x 4 chunks of 8h
      copy_b128_to_lds(Vp + (size_t)d * SEQ + j0 + cv * 8,
                       &vbuf[buf][d][cv * 8]);
    }
  };

  const float scale = 0.08838834764831845f;  // 1/sqrt(128)

  stage(0, 0);
  wait_async0();
  __syncthreads();

  for (int j0 = 0; j0 < SEQ; j0 += 32) {
    const int cur = (j0 >> 5) & 1;
    const int j1 = j0 + 32;
    if (j1 < SEQ) stage(cur ^ 1, j1);           // prefetch next tile (async)

    v8f s0, s1;
#pragma unroll
    for (int j = 0; j < 8; ++j) { s0[j] = 0.f; s1[j] = 0.f; }
#pragma unroll
    for (int dk = 0; dk < 4; ++dk) {
      v16h k0f = load_frag(&kbuf[cur][0][dk * 32], 128);
      v16h k1f = load_frag(&kbuf[cur][16][dk * 32], 128);
      s0 = wmma16(qf[dk], k0f, s0);
      s1 = wmma16(qf[dk], k1f, s1);
    }
#pragma unroll
    for (int j = 0; j < 8; ++j) {
      float a = s0[j] * scale;
      float c = s1[j] * scale;
      float mx = fmaxf(a, c);
#pragma unroll
      for (int msk = 1; msk <= 8; msk <<= 1) mx = fmaxf(mx, __shfl_xor(mx, msk, 32));
      float mnew = fmaxf(mrow[j], mx);
      float fct = __expf(mrow[j] - mnew);
      mrow[j] = mnew;
      float p0 = __expf(a - mnew);
      float p1 = __expf(c - mnew);
      lrow[j] = lrow[j] * fct + p0 + p1;
#pragma unroll
      for (int t = 0; t < 8; ++t) o[t][j] *= fct;
      const int r = rb + j;
      pbuf[wave][r * 32 + n]      = (_Float16)p0;
      pbuf[wave][r * 32 + 16 + n] = (_Float16)p1;
    }
    asm volatile("s_wait_dscnt 0" ::: "memory");  // wave-private P round-trip
    v16h pa = load_frag(&pbuf[wave][0], 32);
#pragma unroll
    for (int t = 0; t < 8; ++t) {
      v16h vf = load_frag(&vbuf[cur][t * 16][0], 32);
      o[t] = wmma16(pa, vf, o[t]);
    }
    wait_async0();     // next tile landed in LDS (this wave's share)
    __syncthreads();   // all waves done staging + done reading `cur`
  }

#pragma unroll
  for (int j = 0; j < 8; ++j) {
    float t = lrow[j];
#pragma unroll
    for (int msk = 1; msk <= 8; msk <<= 1) t += __shfl_xor(t, msk, 32);
    lrow[j] = 1.f / t;
  }
  _Float16* Op = O + base + (size_t)(qb * 64 + wave * 16) * DIM + h * HD;
#pragma unroll
  for (int t = 0; t < 8; ++t)
#pragma unroll
    for (int j = 0; j < 8; ++j)
      Op[(size_t)(rb + j) * DIM + t * 16 + n] = (_Float16)(o[t][j] * lrow[j]);
}

// ---------------------------------------------------------------------------
// Workspace layout (bytes)
// ---------------------------------------------------------------------------
static constexpr size_t SZ_HC   = (size_t)2 * CLEN * DIM * 2;
static constexpr size_t SZ_HX   = (size_t)2 * XLEN * DIM * 2;
static constexpr size_t SZ_W16  = (size_t)DIM * DIM * 2;
static constexpr size_t SZ_F32  = (size_t)2 * SEQ * DIM * 4;
static constexpr size_t SZ_F16  = (size_t)2 * SEQ * DIM * 2;

static constexpr size_t OFF_HC   = 0;
static constexpr size_t OFF_HX   = OFF_HC + SZ_HC;
static constexpr size_t OFF_WT   = OFF_HX + SZ_HX;                  // 8 matrices
static constexpr size_t OFF_QF   = OFF_WT + 8 * SZ_W16;
static constexpr size_t OFF_KF   = OFF_QF + SZ_F32;
static constexpr size_t OFF_VF   = OFF_KF + SZ_F32;
static constexpr size_t OFF_Q16  = OFF_VF + SZ_F32;
static constexpr size_t OFF_K16  = OFF_Q16 + SZ_F16;
static constexpr size_t OFF_VT16 = OFF_K16 + SZ_F16;
static constexpr size_t OFF_ATT  = OFF_QF;   // alias: q fp32 dead after ln_rope

extern "C" void kernel_launch(void* const* d_in, const int* in_sizes, int n_in,
                              void* d_out, int out_size, void* d_ws, size_t ws_size,
                              hipStream_t stream) {
  const float* c    = (const float*)d_in[0];
  const float* x    = (const float*)d_in[1];
  const float* fre  = (const float*)d_in[2];
  const float* fim  = (const float*)d_in[3];
  const float* W[8];
  for (int i = 0; i < 8; ++i) W[i] = (const float*)d_in[4 + i];  // Wq Wk Wv Wo W2q W2k W2v W2o
  const float* qn_w  = (const float*)d_in[12];
  const float* kn_w  = (const float*)d_in[13];
  const float* q2n_w = (const float*)d_in[14];
  const float* k2n_w = (const float*)d_in[15];
  const float* qn_b  = (const float*)d_in[16];
  const float* kn_b  = (const float*)d_in[17];
  const float* q2n_b = (const float*)d_in[18];
  const float* k2n_b = (const float*)d_in[19];

  char* ws = (char*)d_ws;
  _Float16* hc   = (_Float16*)(ws + OFF_HC);
  _Float16* hx   = (_Float16*)(ws + OFF_HX);
  _Float16* wt   = (_Float16*)(ws + OFF_WT);
  float*    qf   = (float*)(ws + OFF_QF);
  float*    kf   = (float*)(ws + OFF_KF);
  float*    vf   = (float*)(ws + OFF_VF);
  _Float16* q16  = (_Float16*)(ws + OFF_Q16);
  _Float16* k16  = (_Float16*)(ws + OFF_K16);
  _Float16* vt16 = (_Float16*)(ws + OFF_VT16);
  _Float16* att  = (_Float16*)(ws + OFF_ATT);

  // 1) convert tokens and weights to f16 (weights transposed)
  {
    int nc = 2 * CLEN * DIM, nx = 2 * XLEN * DIM;
    f32_to_f16_kernel<<<(nc + 255) / 256, 256, 0, stream>>>(c, hc, nc);
    f32_to_f16_kernel<<<(nx + 255) / 256, 256, 0, stream>>>(x, hx, nx);
    for (int i = 0; i < 8; ++i)
      weight_T_kernel<<<dim3(DIM / 32, DIM / 32), dim3(32, 8), 0, stream>>>(
          W[i], wt + (size_t)i * DIM * DIM);
  }

  _Float16* WTq  = wt + 0 * (size_t)DIM * DIM;
  _Float16* WTk  = wt + 1 * (size_t)DIM * DIM;
  _Float16* WTv  = wt + 2 * (size_t)DIM * DIM;
  _Float16* WTo  = wt + 3 * (size_t)DIM * DIM;
  _Float16* WT2q = wt + 4 * (size_t)DIM * DIM;
  _Float16* WT2k = wt + 5 * (size_t)DIM * DIM;
  _Float16* WT2v = wt + 6 * (size_t)DIM * DIM;
  _Float16* WT2o = wt + 7 * (size_t)DIM * DIM;

  // 2) projection GEMMs into fp32 q/k/v buffers laid out [b, 2048, 2048]
  for (int b = 0; b < 2; ++b) {
    const _Float16* Ac = hc + (size_t)b * CLEN * DIM;
    const _Float16* Ax = hx + (size_t)b * XLEN * DIM;
    float* qc = qf + ((size_t)b * SEQ) * DIM;
    float* kc = kf + ((size_t)b * SEQ) * DIM;
    float* vc = vf + ((size_t)b * SEQ) * DIM;
    float* qx = qc + (size_t)CLEN * DIM;
    float* kx = kc + (size_t)CLEN * DIM;
    float* vx = vc + (size_t)CLEN * DIM;
    gemm_f16_kernel<<<dim3(CLEN / 16, DIM / 64), 32, 0, stream>>>(Ac, WTq,  qc, CLEN);
    gemm_f16_kernel<<<dim3(CLEN / 16, DIM / 64), 32, 0, stream>>>(Ac, WTk,  kc, CLEN);
    gemm_f16_kernel<<<dim3(CLEN / 16, DIM / 64), 32, 0, stream>>>(Ac, WTv,  vc, CLEN);
    gemm_f16_kernel<<<dim3(XLEN / 16, DIM / 64), 32, 0, stream>>>(Ax, WT2q, qx, XLEN);
    gemm_f16_kernel<<<dim3(XLEN / 16, DIM / 64), 32, 0, stream>>>(Ax, WT2k, kx, XLEN);
    gemm_f16_kernel<<<dim3(XLEN / 16, DIM / 64), 32, 0, stream>>>(Ax, WT2v, vx, XLEN);
  }

  // 3) LayerNorm + RoPE (fp32 math) -> f16 ; V -> f16 transposed [b,h,d,s]
  ln_rope_kernel<<<2 * SEQ, 256, 0, stream>>>(qf, qn_w, qn_b, q2n_w, q2n_b, fre, fim, q16);
  ln_rope_kernel<<<2 * SEQ, 256, 0, stream>>>(kf, kn_w, kn_b, k2n_w, k2n_b, fre, fim, k16);
  v_transpose_kernel<<<(2 * SEQ * DIM) / 256, 256, 0, stream>>>(vf, vt16);

  // 4) flash attention (overwrites aliased region of qf)
  attn_kernel<<<dim3(SEQ / 64, HEADS, 2), 128, 0, stream>>>(q16, k16, vt16, att);

  // 5) output projections straight into d_out (c_out then x_out, flat)
  float* cout = (float*)d_out;
  float* xout = cout + (size_t)2 * CLEN * DIM;
  for (int b = 0; b < 2; ++b) {
    const _Float16* Ac = att + ((size_t)b * SEQ) * DIM;
    const _Float16* Ax = Ac + (size_t)CLEN * DIM;
    gemm_f16_kernel<<<dim3(CLEN / 16, DIM / 64), 32, 0, stream>>>(
        Ac, WTo, cout + (size_t)b * CLEN * DIM, CLEN);
    gemm_f16_kernel<<<dim3(XLEN / 16, DIM / 64), 32, 0, stream>>>(
        Ax, WT2o, xout + (size_t)b * XLEN * DIM, XLEN);
  }
}
